// JointNetwork_71176198029519
// MI455X (gfx1250) — matrix operand order repaired and checked by reading, not compiled, not run on hardware
//
#include <hip/hip_runtime.h>

#define DEV __device__ __forceinline__

typedef __attribute__((ext_vector_type(8)))  __bf16 v8bf;
typedef __attribute__((ext_vector_type(16))) __bf16 v16bf;
typedef __attribute__((ext_vector_type(8)))  float  v8f;

static constexpr int S_ = 512, B_ = 8, T_ = 64, D_ = 1024, H_ = 8, HD_ = 128,
                     HID_ = 2048, G_ = 16, V_ = 5000, DSZ_ = 32, NL_ = 6;
static constexpr int MF_ = G_ * T_ * B_;   // 8192 rows of x after layer 0
static constexpr int LDP_ = 516;           // padded LDS row stride (floats)

DEV __bf16 f2bf(float f) {
  unsigned u = __builtin_bit_cast(unsigned, f);
  unsigned r = (u + 0x7FFFu + ((u >> 16) & 1u)) >> 16;
  unsigned short h = (unsigned short)r;
  return __builtin_bit_cast(__bf16, h);
}

DEV v8f wmma_bf16(v16bf a, v16bf b, v8f c) {
  // D = A(16x32) * B(32x16) + C, fp32 accumulate
  return __builtin_amdgcn_wmma_f32_16x16x32_bf16(false, a, false, b, (short)0, c,
                                                 false, false);
}

// ---- gfx1250 async global->LDS copy (ASYNCcnt) ----
DEV void async_cp16(unsigned ldsaddr, unsigned long long gbase, unsigned goff) {
  asm volatile("global_load_async_to_lds_b128 %0, %1, %2"
               :
               : "v"(ldsaddr), "v"(goff), "s"(gbase)
               : "memory");
}
DEV void wait_async0() { asm volatile("s_wait_asynccnt 0x0" ::: "memory"); }

// A operand: 16x32 bf16 tile. ISA layout: lanes 0-15 = rows, K packed
// {0..7,16..23}; lanes 16-31 same rows, K {8..15,24..31}.
DEV v16bf load_a_g(const __bf16* A, int lda, int kk) {
  int lane = threadIdx.x & 31;
  const __bf16* p = A + (size_t)(lane & 15) * lda + kk + ((lane & 16) ? 8 : 0);
  v8bf lo = *(const v8bf*)p;
  v8bf hi = *(const v8bf*)(p + 16);
  v16bf r;
#pragma unroll
  for (int i = 0; i < 8; ++i) { r[i] = lo[i]; r[8 + i] = hi[i]; }
  return r;
}

// B operand: 32x16 bf16 tile from W stored (N x K) row-major. Lane n holds
// column n = W row n; lanes 0-15 K {0..7,16..23}, lanes 16-31 K {8..15,24..31}.
DEV v16bf load_b_g(const __bf16* W, int ldb, int n0, int kk, int N) {
  int lane = threadIdx.x & 31;
  int n = n0 + (lane & 15);
  if (n >= N) n = 0;  // clamped read; store is guarded
  const __bf16* p = W + (size_t)n * ldb + kk + ((lane & 16) ? 8 : 0);
  v8bf lo = *(const v8bf*)p;
  v8bf hi = *(const v8bf*)(p + 16);
  v16bf r;
#pragma unroll
  for (int i = 0; i < 8; ++i) { r[i] = lo[i]; r[8 + i] = hi[i]; }
  return r;
}

// Same operand layouts, sourced from LDS-staged tiles (row stride 32 elems)
DEV v16bf load_a_s(const __bf16* Abuf, int m0) {
  int lane = threadIdx.x & 31;
  const __bf16* p = Abuf + (m0 + (lane & 15)) * 32 + ((lane & 16) ? 8 : 0);
  v8bf lo = *(const v8bf*)p;
  v8bf hi = *(const v8bf*)(p + 16);
  v16bf r;
#pragma unroll
  for (int i = 0; i < 8; ++i) { r[i] = lo[i]; r[8 + i] = hi[i]; }
  return r;
}
DEV v16bf load_b_s(const __bf16* Bbuf, int n0) {
  int lane = threadIdx.x & 31;
  const __bf16* p = Bbuf + (n0 + (lane & 15)) * 32 + ((lane & 16) ? 8 : 0);
  v8bf lo = *(const v8bf*)p;
  v8bf hi = *(const v8bf*)(p + 16);
  v16bf r;
#pragma unroll
  for (int i = 0; i < 8; ++i) { r[i] = lo[i]; r[8 + i] = hi[i]; }
  return r;
}

// A operand sourced from fp32 scores in LDS (softmax probs), cvt to bf16.
DEV v16bf load_a_lds(const float* P, int ldp, int m0, int kk) {
  int lane = threadIdx.x & 31;
  const float* p = P + (size_t)(m0 + (lane & 15)) * ldp + kk + ((lane & 16) ? 8 : 0);
  v16bf r;
#pragma unroll
  for (int i = 0; i < 8; ++i) { r[i] = f2bf(p[i]); r[8 + i] = f2bf(p[16 + i]); }
  return r;
}

// C/D layout: VGPR r -> row r (lanes 0-15) / row r+8 (lanes 16-31), col = lane&15
// OMODE: 0 = f32 row-major, 1 = bf16 row-major,
//        2 = bf16 V-transpose  (row=(s,b), col=(h,hd) -> [bh][hd][s]),
//        3 = f32 logits permute (row=(g,t,b) -> orow=((b*G+g)*T+t))
template <int OMODE, int ACT>
DEV void store_tile(const v8f& c, int mb, int nb, const float* bias, float alpha,
                    void* Out, int ldo, int M, int N, int p0, int p1, int p2) {
  int lane = threadIdx.x & 31;
  int col = nb + (lane & 15);
  if (col >= N) return;
  float bv = bias ? bias[col] : 0.f;
  int rb = mb + ((lane & 16) ? 8 : 0);
#pragma unroll
  for (int r = 0; r < 8; ++r) {
    int row = rb + r;
    if (row >= M) continue;
    float v = (c[r] + bv) * alpha;
    if (ACT == 1) v = tanhf(v);
    if (OMODE == 0) {
      ((float*)Out)[(size_t)row * ldo + col] = v;
    } else if (OMODE == 1) {
      ((__bf16*)Out)[(size_t)row * ldo + col] = f2bf(v);
    } else if (OMODE == 2) {
      int b = row % p0, s = row / p0;   // p0 = B
      int h = col / p1, hd = col % p1;  // p1 = HD, p2 = H
      int Srows = M / p0;
      ((__bf16*)Out)[(((size_t)(b * p2 + h)) * p1 + hd) * Srows + s] = f2bf(v);
    } else {
      int b = row % p0;                 // p0 = B, p1 = T, p2 = G
      int gt = row / p0;
      int t = gt % p1, g = gt / p1;
      ((float*)Out)[(((size_t)(b * p2 + g)) * p1 + t) * ldo + col] = v;
    }
  }
}

// C(MxN) = act( A(MxK) * W(NxK)^T + bias ) * alpha
// 8 waves / 256 threads; block tile 128(M) x 64(N); K-slab 32, double-buffered
// in LDS, filled with gfx1250 async global->LDS DMA (ASYNCcnt).
template <int OMODE, int ACT>
__global__ __launch_bounds__(256) void gemm_kernel(
    const __bf16* __restrict__ A, int lda, const __bf16* __restrict__ W, int ldb,
    const float* __restrict__ bias, float alpha, void* __restrict__ Out, int ldo,
    int M, int N, int K, int p0, int p1, int p2) {
  __shared__ __align__(64) __bf16 As[2][128 * 32];  // 2 x 8KB
  __shared__ __align__(64) __bf16 Bs[2][64 * 32];   // 2 x 4KB
  const int tid = threadIdx.x;
  const int wave = tid >> 5;
  const int wm = wave >> 1, wn = wave & 1;
  const int mB = blockIdx.y * 128;
  const int nB = blockIdx.x * 64;

  // each thread moves one 16B granule per async instruction; the LDS tile is
  // contiguous row-major (row = 64B), so dest byte offset is simply tid*16.
  const unsigned grow = tid >> 2;          // granule row 0..63
  const unsigned gcb = (tid & 3) * 16;     // byte column within 64B row
  const unsigned aoff0 = (unsigned)(mB + grow) * (unsigned)(lda * 2) + gcb;
  const unsigned aoff1 = (unsigned)(mB + 64 + grow) * (unsigned)(lda * 2) + gcb;
  const unsigned boff = (unsigned)(nB + grow) * (unsigned)(ldb * 2) + gcb;
  const unsigned long long Ab = (unsigned long long)A;
  const unsigned long long Wb = (unsigned long long)W;
  const unsigned t16 = (unsigned)tid * 16u;
  const unsigned ldsA0 = (unsigned)(size_t)&As[0][0];
  const unsigned ldsA1 = (unsigned)(size_t)&As[1][0];
  const unsigned ldsB0 = (unsigned)(size_t)&Bs[0][0];
  const unsigned ldsB1 = (unsigned)(size_t)&Bs[1][0];

  // prologue: stage K-slab 0 into buffer 0
  async_cp16(ldsA0 + t16, Ab, aoff0);
  async_cp16(ldsA0 + t16 + 4096u, Ab, aoff1);
  async_cp16(ldsB0 + t16, Wb, boff);

  v8f c00 = {}, c01 = {}, c10 = {}, c11 = {};
  int buf = 0;
  for (int kk = 0; kk < K; kk += 32) {
    wait_async0();       // own wave's slab-DMA for `buf` complete
    __syncthreads();     // all waves' slab-DMA visible in LDS
    if (kk + 32 < K) {   // overlap next slab DMA with this slab's WMMA
      unsigned kb = (unsigned)(kk + 32) * 2u;
      unsigned la = buf ? ldsA0 : ldsA1;
      unsigned lb = buf ? ldsB0 : ldsB1;
      async_cp16(la + t16, Ab, aoff0 + kb);
      async_cp16(la + t16 + 4096u, Ab, aoff1 + kb);
      async_cp16(lb + t16, Wb, boff + kb);
    }
    v16bf a0 = load_a_s(&As[buf][0], wm * 32);
    v16bf a1 = load_a_s(&As[buf][0], wm * 32 + 16);
    v16bf b0 = load_b_s(&Bs[buf][0], wn * 32);
    v16bf b1 = load_b_s(&Bs[buf][0], wn * 32 + 16);
    c00 = wmma_bf16(a0, b0, c00);
    c01 = wmma_bf16(a0, b1, c01);
    c10 = wmma_bf16(a1, b0, c10);
    c11 = wmma_bf16(a1, b1, c11);
    __syncthreads();     // reads of `buf` done before it is refilled next iter
    buf ^= 1;
  }
  const int m0 = mB + wm * 32;
  const int n0 = nB + wn * 32;
  store_tile<OMODE, ACT>(c00, m0, n0, bias, alpha, Out, ldo, M, N, p0, p1, p2);
  store_tile<OMODE, ACT>(c01, m0, n0 + 16, bias, alpha, Out, ldo, M, N, p0, p1, p2);
  store_tile<OMODE, ACT>(c10, m0 + 16, n0, bias, alpha, Out, ldo, M, N, p0, p1, p2);
  store_tile<OMODE, ACT>(c11, m0 + 16, n0 + 16, bias, alpha, Out, ldo, M, N, p0, p1, p2);
}

// Fused attention: one block per (group g, head-batch bh). Scores (64x512) via
// WMMA into LDS, masked softmax in LDS, then P@V via WMMA (A from LDS).
__global__ __launch_bounds__(256) void attn_kernel(
    const __bf16* __restrict__ Q, const __bf16* __restrict__ Kb,
    const __bf16* __restrict__ Vt, const unsigned char* __restrict__ kpm,
    __bf16* __restrict__ AO, int g0bcast) {
  extern __shared__ float sc[];  // 64 x LDP_ fp32
  const int g = blockIdx.x;
  const int bh = blockIdx.y;
  const int bb = bh >> 3;  // b  (H = 8)
  const int hh = bh & 7;   // h
  const int lane = threadIdx.x & 31;
  const int wave = threadIdx.x >> 5;
  const int g0 = g0bcast ? 0 : g;
  const int ldr = B_ * D_;  // row stride between t (or s) rows

  const __bf16* Qbase = Q + ((size_t)(g0 * T_) * B_ + bb) * D_ + hh * HD_;
  const __bf16* Kbase = Kb + (size_t)bb * D_ + hh * HD_;

  // ---- phase 1: scores = Q(64x128) . K(512x128)^T -> LDS (fp32) ----
  for (int it = 0; it < 4; ++it) {
    int tid = wave + it * 8;  // 0..31 tiles of 32x32 over (64 x 512)
    int tm = tid >> 4, tn = tid & 15;
    int mrow = tm * 32, ncol = tn * 32;
    v8f c00 = {}, c01 = {}, c10 = {}, c11 = {};
    for (int kk = 0; kk < HD_; kk += 32) {
      v16bf a0 = load_a_g(Qbase + (size_t)mrow * ldr, ldr, kk);
      v16bf a1 = load_a_g(Qbase + (size_t)(mrow + 16) * ldr, ldr, kk);
      v16bf b0 = load_b_g(Kbase + (size_t)ncol * ldr, ldr, 0, kk, 1 << 30);
      v16bf b1 = load_b_g(Kbase + (size_t)(ncol + 16) * ldr, ldr, 0, kk, 1 << 30);
      c00 = wmma_bf16(a0, b0, c00);
      c01 = wmma_bf16(a0, b1, c01);
      c10 = wmma_bf16(a1, b0, c10);
      c11 = wmma_bf16(a1, b1, c11);
    }
    int cc = lane & 15;
#pragma unroll
    for (int r = 0; r < 8; ++r) {
      int rr = r + ((lane & 16) ? 8 : 0);
      sc[(size_t)(mrow + rr) * LDP_ + ncol + cc] = c00[r];
      sc[(size_t)(mrow + rr) * LDP_ + ncol + 16 + cc] = c01[r];
      sc[(size_t)(mrow + 16 + rr) * LDP_ + ncol + cc] = c10[r];
      sc[(size_t)(mrow + 16 + rr) * LDP_ + ncol + 16 + cc] = c11[r];
    }
  }
  __syncthreads();

  // ---- phase 2: masked softmax over s (rows of 512), in-place in LDS ----
  const int smax = (g + 1) * DSZ_;
  for (int t = wave * 8; t < wave * 8 + 8; ++t) {
    float vbuf[16];
    float mx = -1e30f;
#pragma unroll
    for (int j = 0; j < 16; ++j) {
      int s = lane + j * 32;
      float v = sc[(size_t)t * LDP_ + s];
      bool dead = (s >= smax) || (kpm[bb * S_ + s] != 0);
      v = dead ? -1e30f : v;
      vbuf[j] = v;
      mx = fmaxf(mx, v);
    }
#pragma unroll
    for (int o = 16; o; o >>= 1) mx = fmaxf(mx, __shfl_xor(mx, o));
    float sum = 0.f;
#pragma unroll
    for (int j = 0; j < 16; ++j) {
      float e = (vbuf[j] <= -1e29f) ? 0.f : __expf(vbuf[j] - mx);
      vbuf[j] = e;
      sum += e;
    }
#pragma unroll
    for (int o = 16; o; o >>= 1) sum += __shfl_xor(sum, o);
    float inv = 1.f / sum;
#pragma unroll
    for (int j = 0; j < 16; ++j) sc[(size_t)t * LDP_ + lane + j * 32] = vbuf[j] * inv;
  }
  __syncthreads();

  // ---- phase 3: O(64x128) = P(64x512, LDS) . V(512x128); V^T rows are hd ----
  {
    int tm = wave >> 2, tn = wave & 3;
    int mrow = tm * 32, ncol = tn * 32;
    const __bf16* Vb = Vt + (size_t)bh * HD_ * S_;  // row = hd, ld = S
    v8f c00 = {}, c01 = {}, c10 = {}, c11 = {};
    for (int kk = 0; kk < S_; kk += 32) {
      v16bf a0 = load_a_lds(sc, LDP_, mrow, kk);
      v16bf a1 = load_a_lds(sc, LDP_, mrow + 16, kk);
      v16bf b0 = load_b_g(Vb, S_, ncol, kk, HD_);
      v16bf b1 = load_b_g(Vb, S_, ncol + 16, kk, HD_);
      c00 = wmma_bf16(a0, b0, c00);
      c01 = wmma_bf16(a0, b1, c01);
      c10 = wmma_bf16(a1, b0, c10);
      c11 = wmma_bf16(a1, b1, c11);
    }
    int cc = lane & 15;
    __bf16* Ob = AO + hh * HD_;
#pragma unroll
    for (int r = 0; r < 8; ++r) {
      int rr = r + ((lane & 16) ? 8 : 0);
      size_t r0 = ((size_t)(g * T_ + mrow + rr) * B_ + bb) * D_;
      size_t r1 = ((size_t)(g * T_ + mrow + 16 + rr) * B_ + bb) * D_;
      Ob[r0 + ncol + cc] = f2bf(c00[r]);
      Ob[r0 + ncol + 16 + cc] = f2bf(c01[r]);
      Ob[r1 + ncol + cc] = f2bf(c10[r]);
      Ob[r1 + ncol + 16 + cc] = f2bf(c11[r]);
    }
  }
}

// x_new = LN(t + x_old) ; one block per row of D=1024; writes fp32 + bf16
__global__ __launch_bounds__(256) void add_ln_kernel(
    const float* __restrict__ tin, const float* __restrict__ xo,
    const float* __restrict__ gam, const float* __restrict__ bet,
    float* __restrict__ xn, __bf16* __restrict__ xbf, int rowsOld) {
  __shared__ float rs[8], rs2[8];
  int row = blockIdx.x;
  int orow = row % rowsOld;  // broadcast for layer 0 residual
  float v[4], s = 0.f, s2 = 0.f;
#pragma unroll
  for (int i = 0; i < 4; ++i) {
    int d = threadIdx.x + i * 256;
    float a = tin[(size_t)row * D_ + d] + xo[(size_t)orow * D_ + d];
    v[i] = a;
    s += a;
    s2 += a * a;
  }
#pragma unroll
  for (int o = 16; o; o >>= 1) { s += __shfl_xor(s, o); s2 += __shfl_xor(s2, o); }
  int lane = threadIdx.x & 31, wave = threadIdx.x >> 5;
  if (lane == 0) { rs[wave] = s; rs2[wave] = s2; }
  __syncthreads();
  float ts = 0.f, t2 = 0.f;
#pragma unroll
  for (int w = 0; w < 8; ++w) { ts += rs[w]; t2 += rs2[w]; }
  float mu = ts * (1.f / D_);
  float var = t2 * (1.f / D_) - mu * mu;
  float rinv = rsqrtf(var + 1e-5f);
#pragma unroll
  for (int i = 0; i < 4; ++i) {
    int d = threadIdx.x + i * 256;
    float y = (v[i] - mu) * rinv * gam[d] + bet[d];
    xn[(size_t)row * D_ + d] = y;
    xbf[(size_t)row * D_ + d] = f2bf(y);
  }
}

__global__ void cvt_kernel(const float* __restrict__ in, __bf16* __restrict__ out, int n) {
  int i = blockIdx.x * blockDim.x + threadIdx.x;
  if (i < n) out[i] = f2bf(in[i]);
}

// x[(t*B+b)*D+d] = dec[(b*T+t)*D+d]  (plus bf16 copy)
__global__ void init_x_kernel(const float* __restrict__ dec, float* __restrict__ x,
                              __bf16* __restrict__ xbf) {
  int i = blockIdx.x * blockDim.x + threadIdx.x;
  int n = T_ * B_ * D_;
  if (i >= n) return;
  int d = i % D_;
  int r = i / D_;
  int b = r % B_;
  int t = r / B_;
  float v = dec[((size_t)b * T_ + t) * D_ + d];
  x[i] = v;
  xbf[i] = f2bf(v);
}

__global__ void glen_kernel(const unsigned char* __restrict__ kpm, void* __restrict__ out) {
  int b = threadIdx.x;
  if (b >= B_) return;
  int len = 0;
  for (int s = 0; s < S_; ++s) len += (kpm[b * S_ + s] == 0) ? 1 : 0;
  int gl = (len + DSZ_ - 1) / DSZ_;
  ((int*)out)[(size_t)B_ * G_ * T_ * V_ + b] = gl;
}

extern "C" void kernel_launch(void* const* d_in, const int* in_sizes, int n_in,
                              void* d_out, int out_size, void* d_ws, size_t ws_size,
                              hipStream_t stream) {
  (void)in_sizes; (void)n_in; (void)out_size; (void)ws_size;
  const float* enc = (const float*)d_in[0];
  const unsigned char* kpm = (const unsigned char*)d_in[1];
  const float* dec = (const float*)d_in[2];
  // params: 16 arrays per layer, then lin_out_w, lin_out_b
  auto LP = [&](int L, int idx) -> const float* {
    return (const float*)d_in[3 + L * 16 + idx];
  };
  const float* lo_w = (const float*)d_in[3 + NL_ * 16 + 0];
  const float* lo_b = (const float*)d_in[3 + NL_ * 16 + 1];

  // ---- carve workspace (256B aligned chunks) ----
  size_t off = 0;
  auto carve = [&](size_t bytes) -> void* {
    void* p = (char*)d_ws + off;
    off += (bytes + 255) & ~(size_t)255;
    return p;
  };
  __bf16* enc_bf = (__bf16*)carve((size_t)S_ * B_ * D_ * 2);
  __bf16* lo_bf  = (__bf16*)carve((size_t)5056 * D_ * 2);  // padded to 79*64 rows
  __bf16* wq_bf  = (__bf16*)carve((size_t)D_ * D_ * 2);
  __bf16* wk_bf  = (__bf16*)carve((size_t)D_ * D_ * 2);
  __bf16* wv_bf  = (__bf16*)carve((size_t)D_ * D_ * 2);
  __bf16* wo_bf  = (__bf16*)carve((size_t)D_ * D_ * 2);
  __bf16* w1_bf  = (__bf16*)carve((size_t)HID_ * D_ * 2);
  __bf16* w2_bf  = (__bf16*)carve((size_t)D_ * HID_ * 2);
  float*  xA     = (float*)carve((size_t)MF_ * D_ * 4);
  float*  xB     = (float*)carve((size_t)MF_ * D_ * 4);
  __bf16* x_bf   = (__bf16*)carve((size_t)MF_ * D_ * 2);
  __bf16* q_bf   = (__bf16*)carve((size_t)MF_ * D_ * 2);
  __bf16* k_bf   = (__bf16*)carve((size_t)S_ * B_ * D_ * 2);
  __bf16* vt_bf  = (__bf16*)carve((size_t)S_ * B_ * D_ * 2);
  __bf16* ao_bf  = (__bf16*)carve((size_t)MF_ * D_ * 2);
  __bf16* h_bf   = (__bf16*)carve((size_t)MF_ * HID_ * 2);
  float*  tmp    = (float*)carve((size_t)MF_ * D_ * 4);

  const float SCAL = 0.08838834764831845f;  // HD^-0.5
  auto cgrid = [](int n) { return dim3((n + 255) / 256); };

  cvt_kernel<<<cgrid(S_ * B_ * D_), 256, 0, stream>>>(enc, enc_bf, S_ * B_ * D_);
  cvt_kernel<<<cgrid(V_ * D_), 256, 0, stream>>>(lo_w, lo_bf, V_ * D_);
  init_x_kernel<<<cgrid(T_ * B_ * D_), 256, 0, stream>>>(dec, xA, x_bf);

  float* xcur = xA;
  float* xnxt = xB;
  int Mx = T_ * B_;  // x rows (512 for layer 0, then 8192)
  const size_t attn_lds = (size_t)64 * LDP_ * 4;  // 132 KB (<320 KB/WGP)

  for (int L = 0; L < NL_; ++L) {
    cvt_kernel<<<cgrid(D_ * D_), 256, 0, stream>>>(LP(L, 0), wq_bf, D_ * D_);
    cvt_kernel<<<cgrid(D_ * D_), 256, 0, stream>>>(LP(L, 2), wk_bf, D_ * D_);
    cvt_kernel<<<cgrid(D_ * D_), 256, 0, stream>>>(LP(L, 4), wv_bf, D_ * D_);
    cvt_kernel<<<cgrid(D_ * D_), 256, 0, stream>>>(LP(L, 6), wo_bf, D_ * D_);
    cvt_kernel<<<cgrid(HID_ * D_), 256, 0, stream>>>(LP(L, 8), w1_bf, HID_ * D_);
    cvt_kernel<<<cgrid(D_ * HID_), 256, 0, stream>>>(LP(L, 10), w2_bf, D_ * HID_);

    // q = (x @ wq^T + bq) * scaling  -> bf16 (Mx x D)
    gemm_kernel<1, 0><<<dim3(D_ / 64, Mx / 128), 256, 0, stream>>>(
        x_bf, D_, wq_bf, D_, LP(L, 1), SCAL, q_bf, D_, Mx, D_, D_, 0, 0, 0);
    // k = enc @ wk^T + bk  -> bf16 (S*B x D)
    gemm_kernel<1, 0><<<dim3(D_ / 64, (S_ * B_) / 128), 256, 0, stream>>>(
        enc_bf, D_, wk_bf, D_, LP(L, 3), 1.f, k_bf, D_, S_ * B_, D_, D_, 0, 0, 0);
    // v = enc @ wv^T + bv  -> bf16, stored transposed per head: [bh][hd][s]
    gemm_kernel<2, 0><<<dim3(D_ / 64, (S_ * B_) / 128), 256, 0, stream>>>(
        enc_bf, D_, wv_bf, D_, LP(L, 5), 1.f, vt_bf, 0, S_ * B_, D_, D_, B_, HD_, H_);

    // fused scores + masked softmax + P@V -> ao bf16 (G,T,B,D)
    attn_kernel<<<dim3(G_, B_ * H_), 256, attn_lds, stream>>>(
        q_bf, k_bf, vt_bf, kpm, ao_bf, (L == 0) ? 1 : 0);

    // tmp = ao @ wo^T + bo  (fp32)
    gemm_kernel<0, 0><<<dim3(D_ / 64, MF_ / 128), 256, 0, stream>>>(
        ao_bf, D_, wo_bf, D_, LP(L, 7), 1.f, tmp, D_, MF_, D_, D_, 0, 0, 0);
    // x = LN(tmp + x)   (layer 0: residual broadcasts over groups)
    add_ln_kernel<<<MF_, 256, 0, stream>>>(tmp, xcur, LP(L, 12), LP(L, 13), xnxt,
                                           x_bf, Mx);
    { float* t_ = xcur; xcur = xnxt; xnxt = t_; }
    Mx = MF_;

    // h = tanh(x @ w1^T + b1) -> bf16 (MF x HID)
    gemm_kernel<1, 1><<<dim3(HID_ / 64, MF_ / 128), 256, 0, stream>>>(
        x_bf, D_, w1_bf, D_, LP(L, 9), 1.f, h_bf, HID_, MF_, HID_, D_, 0, 0, 0);
    // tmp = h @ w2^T + b2 (fp32)
    gemm_kernel<0, 0><<<dim3(D_ / 64, MF_ / 128), 256, 0, stream>>>(
        h_bf, HID_, w2_bf, HID_, LP(L, 11), 1.f, tmp, D_, MF_, D_, HID_, 0, 0, 0);
    // x = LN(tmp + x)
    add_ln_kernel<<<MF_, 256, 0, stream>>>(tmp, xcur, LP(L, 14), LP(L, 15), xnxt,
                                           x_bf, MF_);
    { float* t_ = xcur; xcur = xnxt; xnxt = t_; }
  }

  // logits: x(g,t,b) @ lin_out^T + b -> fp32 at out[((b*G+g)*T+t)*V + n]
  gemm_kernel<3, 0><<<dim3((V_ + 63) / 64, MF_ / 128), 256, 0, stream>>>(
      x_bf, D_, lo_bf, D_, lo_b, 1.f, d_out, V_, MF_, V_, D_, B_, T_, G_);
  // group_lengths (int32 bits) appended after logits
  glen_kernel<<<1, 32, 0, stream>>>(kpm, d_out);
}
